// SwinTransformerBlock_12996571037709
// MI455X (gfx1250) — matrix operand order repaired and checked by reading, not compiled, not run on hardware
//
#include <hip/hip_runtime.h>
#include <hip/hip_bf16.h>

// ---------------------------------------------------------------------------
// Swin 3D block, MI455X (gfx1250, wave32, WMMA 16x16x32 f16->f32)
// ---------------------------------------------------------------------------
typedef __attribute__((ext_vector_type(16))) _Float16 v16h;
typedef __attribute__((ext_vector_type(8)))  _Float16 v8h;
typedef __attribute__((ext_vector_type(4)))  _Float16 v4h;
typedef __attribute__((ext_vector_type(8)))  float    v8f;

union HV16 { v16h v; v8h h[2]; };

#define L_TOK   43904      // 14*56*56
#define CDIM    192
#define NTOK    343
#define NPAD    352
#define NWIN    128
#define HEADS   6
#define QKVDIM  576
#define HID     768
#define NTYPE   8          // 8 canonical shifted-window mask patterns
#define SCALE_Q 0.17677669529663687f   // 32^-0.5

// ---- DPP16 butterfly reductions (no LDS, pure VALU) -----------------------
template <int CTRL>
__device__ __forceinline__ float dppf(float x) {
  union { float f; int i; } u; u.f = x;
  u.i = __builtin_amdgcn_update_dpp(u.i, u.i, CTRL, 0xF, 0xF, true);
  return u.f;
}
__device__ __forceinline__ float red_max16(float x) {
  x = fmaxf(x, dppf<0xB1>(x));    // quad_perm(1,0,3,2)  : xor 1
  x = fmaxf(x, dppf<0x4E>(x));    // quad_perm(2,3,0,1)  : xor 2
  x = fmaxf(x, dppf<0x141>(x));   // ROW_HALF_MIRROR     : xor 4
  x = fmaxf(x, dppf<0x140>(x));   // ROW_MIRROR          : xor 8
  return x;
}
__device__ __forceinline__ float red_sum16(float x) {
  x += dppf<0xB1>(x);
  x += dppf<0x4E>(x);
  x += dppf<0x141>(x);
  x += dppf<0x140>(x);
  return x;
}

// map window-layout row index -> flat token index l (handles roll by -3 / +3)
__device__ __forceinline__ int win_to_l(int rw) {
  int win = rw / NTOK, t = rw % NTOK;
  int ws = win >> 6, wh = (win >> 3) & 7, ww = win & 7;
  int ts = t / 49, rr = t % 49, th = rr / 7, tw = rr % 7;
  int s = ws * 7 + ts + 3; if (s >= 14) s -= 14;
  int h = wh * 7 + th + 3; if (h >= 56) h -= 56;
  int w = ww * 7 + tw + 3; if (w >= 56) w -= 56;
  return (s * 56 + h) * 56 + w;
}

// ---------------------------------------------------------------------------
// Weight convert f32->f16 with transpose: Wt[n][k] = W[k][n]
// ---------------------------------------------------------------------------
__global__ __launch_bounds__(256) void k_cvtw(
    const float* __restrict__ qkv_w, const float* __restrict__ proj_w,
    const float* __restrict__ fc1_w, const float* __restrict__ fc2_w,
    _Float16* __restrict__ WtQ, _Float16* __restrict__ WtP,
    _Float16* __restrict__ WtF1, _Float16* __restrict__ WtF2) {
  int i = blockIdx.x * 256 + threadIdx.x;
  if (i < 110592) {                       // qkv: 576 x 192
    int n = i / 192, k = i % 192;
    WtQ[i] = (_Float16)qkv_w[k * 576 + n];
  } else if (i < 147456) {                // proj: 192 x 192
    int j = i - 110592; int n = j / 192, k = j % 192;
    WtP[j] = (_Float16)proj_w[k * 192 + n];
  } else if (i < 294912) {                // fc1: 768 x 192
    int j = i - 147456; int n = j / 192, k = j % 192;
    WtF1[j] = (_Float16)fc1_w[k * 768 + n];
  } else if (i < 442368) {                // fc2: 192 x 768
    int j = i - 294912; int n = j / 768, k = j % 768;
    WtF2[j] = (_Float16)fc2_w[k * 192 + n];
  }
}

// ---------------------------------------------------------------------------
// Fused rel-pos bias + window mask, padded: BiasM[type][head][352][352].
// type bits: 4 = S-mixed (ws==1), 2 = H-mixed (wh==7), 1 = W-mixed (ww==7);
// in a mixed dimension the region splits at local coordinate 4.
// Padded key cols get -1e30.
// ---------------------------------------------------------------------------
__global__ __launch_bounds__(256) void k_bias(const float* __restrict__ rpb,
                                              float* __restrict__ BiasM) {
  int i = blockIdx.x * 256 + threadIdx.x;   // < 8*6*352*352
  int type = i / (HEADS * NPAD * NPAD);
  int rem  = i % (HEADS * NPAD * NPAD);
  int h = rem / (NPAD * NPAD);
  rem = rem % (NPAD * NPAD);
  int m = rem / NPAD, n = rem % NPAD;
  float val;
  if (n >= NTOK)       val = -1e30f;
  else if (m >= NTOK)  val = 0.f;
  else {
    int ms = m / 49, mr = m % 49, mh = mr / 7, mw = mr % 7;
    int ns = n / 49, nr = n % 49, nh = nr / 7, nw = nr % 7;
    int idx = ((ms - ns + 6) * 13 + (mh - nh + 6)) * 13 + (mw - nw + 6);
    val = rpb[idx * HEADS + h];
    int bs = (type >> 2) & 1, bh = (type >> 1) & 1, bw = type & 1;
    int regm = (bs & (ms >= 4)) * 4 + (bh & (mh >= 4)) * 2 + (bw & (mw >= 4));
    int regn = (bs & (ns >= 4)) * 4 + (bh & (nh >= 4)) * 2 + (bw & (nw >= 4));
    if (regm != regn) val += -100.f;
  }
  BiasM[i] = val;
}

// ---------------------------------------------------------------------------
// LN1 + roll + window partition.  One warp per windowed token.
// Writes Xw (window layout, f16) and xt shortcut (L,C f32).
// ---------------------------------------------------------------------------
__global__ __launch_bounds__(256) void k_ln1(
    const float* __restrict__ x, const float* __restrict__ g,
    const float* __restrict__ b, _Float16* __restrict__ Xw,
    float* __restrict__ xt) {
  int wave = threadIdx.x >> 5, lane = threadIdx.x & 31;
  int rw = blockIdx.x * 8 + wave;
  int l = win_to_l(rw);
  float v[6], s = 0.f, sq = 0.f;
#pragma unroll
  for (int j = 0; j < 6; j++) {
    int c = lane + 32 * j;
    float t = x[(size_t)c * L_TOK + l];
    v[j] = t; s += t; sq += t * t;
  }
#pragma unroll
  for (int off = 16; off >= 1; off >>= 1) {
    s  += __shfl_xor(s,  off, 32);
    sq += __shfl_xor(sq, off, 32);
  }
  float mean = s * (1.f / 192.f);
  float var  = sq * (1.f / 192.f) - mean * mean;
  float rstd = rsqrtf(var + 1e-5f);
#pragma unroll
  for (int j = 0; j < 6; j++) {
    int c = lane + 32 * j;
    xt[(size_t)l * CDIM + c] = v[j];
    Xw[(size_t)rw * CDIM + c] = (_Float16)((v[j] - mean) * rstd * g[c] + b[c]);
  }
}

// LN2 over xt2 (L,C f32) -> Xh f16
__global__ __launch_bounds__(256) void k_ln2(
    const float* __restrict__ xt2, const float* __restrict__ g,
    const float* __restrict__ b, _Float16* __restrict__ Xh) {
  int wave = threadIdx.x >> 5, lane = threadIdx.x & 31;
  int l = blockIdx.x * 8 + wave;
  float v[6], s = 0.f, sq = 0.f;
#pragma unroll
  for (int j = 0; j < 6; j++) {
    int c = lane + 32 * j;
    float t = xt2[(size_t)l * CDIM + c];
    v[j] = t; s += t; sq += t * t;
  }
#pragma unroll
  for (int off = 16; off >= 1; off >>= 1) {
    s  += __shfl_xor(s,  off, 32);
    sq += __shfl_xor(sq, off, 32);
  }
  float mean = s * (1.f / 192.f);
  float var  = sq * (1.f / 192.f) - mean * mean;
  float rstd = rsqrtf(var + 1e-5f);
#pragma unroll
  for (int j = 0; j < 6; j++) {
    int c = lane + 32 * j;
    Xh[(size_t)l * CDIM + c] = (_Float16)((v[j] - mean) * rstd * g[c] + b[c]);
  }
}

// ---------------------------------------------------------------------------
// Generic WMMA GEMM: out(M x Nout) = A(M x K, f16 rowmajor) * Wt^T + bias
// Wt stored transposed (Nout x K).  Block = 8 waves -> 128x64 tile.
// MODE 0: qkv (scale q cols, f16 out)     MODE 1: proj (+window reverse+res)
// MODE 2: fc1 (+GELU, f16 out)            MODE 3: fc2 (+res, transpose store)
// ---------------------------------------------------------------------------
template <int MODE>
__global__ __launch_bounds__(256) void k_gemm(
    const _Float16* __restrict__ A, const _Float16* __restrict__ Wt,
    const float* __restrict__ bias, int K, int Nout,
    _Float16* __restrict__ outH, float* __restrict__ outF,
    const float* __restrict__ res) {
  int wave = threadIdx.x >> 5, lane = threadIdx.x & 31;
  int laneLo = lane & 15, laneHi = lane >> 4;
  int m0 = blockIdx.y * 128 + wave * 16;
  int n0 = blockIdx.x * 64;

  v8f acc[4] = {v8f{}, v8f{}, v8f{}, v8f{}};
  const _Float16* arow = A + (size_t)(m0 + laneLo) * K;

  for (int kk = 0; kk < K; kk += 32) {
    __builtin_prefetch((const void*)(arow + kk + 256), 0, 3);
    HV16 a;
    a.h[0] = *(const v8h*)(arow + kk + laneHi * 16);
    a.h[1] = *(const v8h*)(arow + kk + laneHi * 16 + 8);
#pragma unroll
    for (int t = 0; t < 4; t++) {
      const _Float16* wp = Wt + (size_t)(n0 + t * 16 + laneLo) * K + kk + laneHi * 16;
      HV16 bfrag;
      bfrag.h[0] = *(const v8h*)(wp);
      bfrag.h[1] = *(const v8h*)(wp + 8);
      acc[t] = __builtin_amdgcn_wmma_f32_16x16x32_f16(
          false, a.v, false, bfrag.v, (short)0, acc[t], false, false);
    }
  }

  int lidx[8];
#pragma unroll
  for (int r = 0; r < 8; r++) {
    int m = m0 + r + 8 * laneHi;
    lidx[r] = (MODE == 1) ? win_to_l(m) : m;
  }

#pragma unroll
  for (int t = 0; t < 4; t++) {
    int c = n0 + t * 16 + laneLo;
    float bv = bias[c];
#pragma unroll
    for (int r = 0; r < 8; r++) {
      int m = m0 + r + 8 * laneHi;
      float v = acc[t][r] + bv;
      if (MODE == 0) {
        if (c < CDIM) v *= SCALE_Q;
        outH[(size_t)m * Nout + c] = (_Float16)v;
      } else if (MODE == 1) {
        int l = lidx[r];
        outF[(size_t)l * CDIM + c] = v + res[(size_t)l * CDIM + c];
      } else if (MODE == 2) {
        float gl = 0.5f * v * (1.0f + erff(v * 0.70710678118654752f));
        outH[(size_t)m * Nout + c] = (_Float16)gl;
      } else {  // MODE 3: final residual + transpose store (C, L)
        float fv = v + res[(size_t)m * CDIM + c];
        outF[(size_t)c * L_TOK + m] = fv;
      }
    }
  }
}

// ---------------------------------------------------------------------------
// Flash attention per (window, head): 8 waves, row tiles of 16 queries.
// ALL of K (rowmajor) and V (transposed) for the head are staged into LDS
// once (50 KB << 320 KB/WGP); bias+mask come fused from BiasM (one load+add
// per score element); softmax reductions are DPP16.  Barrier-free main loop.
// ---------------------------------------------------------------------------
__global__ __launch_bounds__(256) void k_attn(
    const _Float16* __restrict__ QKV, const float* __restrict__ BiasM,
    _Float16* __restrict__ O) {
  __shared__ _Float16 Kt[NPAD][40];        // [token][dim]      28160 B
  __shared__ _Float16 VtT[32][NPAD + 8];   // [dim][token]      23040 B
  __shared__ _Float16 Pw[8][16][40];       // per-wave P scratch 10240 B

  int win = blockIdx.x / HEADS, head = blockIdx.x % HEADS;
  int wave = threadIdx.x >> 5, lane = threadIdx.x & 31;
  int laneLo = lane & 15, laneHi = lane >> 4;

  const _Float16* base = QKV + (size_t)win * NTOK * QKVDIM + head * 32;
  const _Float16* Qp = base;
  const _Float16* Kp = base + 192;
  const _Float16* Vp = base + 384;

  // window mask type: S-mixed (ws==1), H-mixed (wh==7), W-mixed (ww==7)
  int type = (((win >> 6) == 1) ? 4 : 0) |
             ((((win >> 3) & 7) == 7) ? 2 : 0) |
             (((win & 7) == 7) ? 1 : 0);
  const float* Bh = BiasM + ((size_t)(type * HEADS + head)) * NPAD * NPAD;

  // ---- one-shot staging: K rowmajor, V transposed -----------------------
  for (int task = threadIdx.x; task < NPAD * 8; task += 256) {
    int tok = task >> 3;                   // 0..351
    int dq  = (task & 7) * 4;              // 0..28
    int tg  = (tok > NTOK - 1) ? (NTOK - 1) : tok;
    *(v4h*)(&Kt[tok][dq]) = *(const v4h*)(Kp + (size_t)tg * QKVDIM + dq);
    v4h vv = *(const v4h*)(Vp + (size_t)tg * QKVDIM + dq);
    VtT[dq + 0][tok] = vv[0]; VtT[dq + 1][tok] = vv[1];
    VtT[dq + 2][tok] = vv[2]; VtT[dq + 3][tok] = vv[3];
  }
  __syncthreads();

  // ---- barrier-free main loop -------------------------------------------
  for (int it = 0; it < 3; ++it) {
    int rt = wave + it * 8;
    bool act = rt < 22;                    // 22 row tiles of 16 cover 343
    if (!act) continue;                    // wave-uniform, no barriers below

    int qrow = rt * 16 + laneLo; if (qrow > NTOK - 1) qrow = NTOK - 1;
    const _Float16* qp = Qp + (size_t)qrow * QKVDIM;
    HV16 qa;
    qa.h[0] = *(const v8h*)(qp + laneHi * 16);
    qa.h[1] = *(const v8h*)(qp + laneHi * 16 + 8);

    v8f o0 = {}, o1 = {};
    float mrow[8], lrow[8];
    const float* brow[8];
#pragma unroll
    for (int r = 0; r < 8; r++) {
      mrow[r] = -1e30f; lrow[r] = 0.f;
      int q = rt * 16 + r + 8 * laneHi;    // < 352
      brow[r] = Bh + (size_t)q * NPAD;
    }

    for (int jp = 0; jp < 11; jp++) {      // 11 pair-tiles of 32 keys = 352
      v8f s[2];
#pragma unroll
      for (int jj = 0; jj < 2; jj++) {
        int n = jp * 32 + jj * 16 + laneLo;
        HV16 kb;
        kb.h[0] = *(const v8h*)(&Kt[n][laneHi * 16]);
        kb.h[1] = *(const v8h*)(&Kt[n][laneHi * 16 + 8]);
        v8f z = {};
        s[jj] = __builtin_amdgcn_wmma_f32_16x16x32_f16(
            false, qa.v, false, kb.v, (short)0, z, false, false);
#pragma unroll
        for (int r = 0; r < 8; r++) {
          s[jj][r] += brow[r][n];          // fused rel-pos bias + mask
        }
      }
      // online softmax over the 32 new key columns (DPP16 reductions)
#pragma unroll
      for (int r = 0; r < 8; r++) {
        float mx = red_max16(fmaxf(s[0][r], s[1][r]));
        float mnew  = fmaxf(mrow[r], mx);
        float alpha = __expf(mrow[r] - mnew);
        float p0 = __expf(s[0][r] - mnew);
        float p1 = __expf(s[1][r] - mnew);
        float sum = red_sum16(p0 + p1);
        lrow[r] = lrow[r] * alpha + sum;
        mrow[r] = mnew;
        o0[r] *= alpha; o1[r] *= alpha;
        int mloc = r + 8 * laneHi;
        Pw[wave][mloc][laneLo]      = (_Float16)p0;
        Pw[wave][mloc][16 + laneLo] = (_Float16)p1;
      }
      // P (A-frag) and V (B-frags) -> accumulate O
      HV16 pa;
      pa.h[0] = *(const v8h*)(&Pw[wave][laneLo][laneHi * 16]);
      pa.h[1] = *(const v8h*)(&Pw[wave][laneLo][laneHi * 16 + 8]);
      int kb0 = jp * 32 + laneHi * 16;
      HV16 vb0, vb1;
      vb0.h[0] = *(const v8h*)(&VtT[laneLo][kb0]);
      vb0.h[1] = *(const v8h*)(&VtT[laneLo][kb0 + 8]);
      vb1.h[0] = *(const v8h*)(&VtT[16 + laneLo][kb0]);
      vb1.h[1] = *(const v8h*)(&VtT[16 + laneLo][kb0 + 8]);
      o0 = __builtin_amdgcn_wmma_f32_16x16x32_f16(
          false, pa.v, false, vb0.v, (short)0, o0, false, false);
      o1 = __builtin_amdgcn_wmma_f32_16x16x32_f16(
          false, pa.v, false, vb1.v, (short)0, o1, false, false);
    }

#pragma unroll
    for (int r = 0; r < 8; r++) {
      int q = rt * 16 + r + 8 * laneHi;
      if (q < NTOK) {
        float inv = 1.0f / lrow[r];
        size_t o = (size_t)(win * NTOK + q) * CDIM + head * 32 + laneLo;
        O[o]      = (_Float16)(o0[r] * inv);
        O[o + 16] = (_Float16)(o1[r] * inv);
      }
    }
  }
}

// ---------------------------------------------------------------------------
extern "C" void kernel_launch(void* const* d_in, const int* in_sizes, int n_in,
                              void* d_out, int out_size, void* d_ws, size_t ws_size,
                              hipStream_t stream) {
  const float* x      = (const float*)d_in[0];
  const float* n1g    = (const float*)d_in[1];
  const float* n1b    = (const float*)d_in[2];
  const float* qkv_w  = (const float*)d_in[3];
  const float* qkv_b  = (const float*)d_in[4];
  const float* rpb    = (const float*)d_in[5];
  const float* proj_w = (const float*)d_in[6];
  const float* proj_b = (const float*)d_in[7];
  const float* n2g    = (const float*)d_in[8];
  const float* n2b    = (const float*)d_in[9];
  const float* fc1_w  = (const float*)d_in[10];
  const float* fc1_b  = (const float*)d_in[11];
  const float* fc2_w  = (const float*)d_in[12];
  const float* fc2_b  = (const float*)d_in[13];
  float* out = (float*)d_out;

  char* w = (char*)d_ws;
  size_t off = 0;
  auto nxt = [&](size_t bytes) -> void* {
    void* p = w + off;
    off += (bytes + 255) & ~(size_t)255;
    return p;
  };
  const size_t L = L_TOK;
  float*    xt    = (float*)   nxt(L * CDIM * 4);
  _Float16* Xw    = (_Float16*)nxt(L * CDIM * 2);
  _Float16* QKVb  = (_Float16*)nxt(L * QKVDIM * 2);
  float*    BiasM = (float*)   nxt((size_t)NTYPE * HEADS * NPAD * NPAD * 4);
  _Float16* AttnO = (_Float16*)nxt(L * CDIM * 2);
  float*    xt2   = (float*)   nxt(L * CDIM * 4);
  _Float16* Xh    = (_Float16*)nxt(L * CDIM * 2);
  _Float16* Hmid  = (_Float16*)nxt(L * HID * 2);
  _Float16* WtQ   = (_Float16*)nxt((size_t)QKVDIM * CDIM * 2);
  _Float16* WtP   = (_Float16*)nxt((size_t)CDIM * CDIM * 2);
  _Float16* WtF1  = (_Float16*)nxt((size_t)HID * CDIM * 2);
  _Float16* WtF2  = (_Float16*)nxt((size_t)CDIM * HID * 2);

  // 1. weights -> f16 transposed
  k_cvtw<<<1728, 256, 0, stream>>>(qkv_w, proj_w, fc1_w, fc2_w,
                                   WtQ, WtP, WtF1, WtF2);
  // 2. fused rel-pos bias + mask tables (8 window types x 6 heads)
  k_bias<<<(NTYPE * HEADS * NPAD * NPAD) / 256, 256, 0, stream>>>(rpb, BiasM);
  // 3. LN1 + roll + window partition (+ shortcut transpose)
  k_ln1<<<L_TOK / 8, 256, 0, stream>>>(x, n1g, n1b, Xw, xt);
  // 4. QKV GEMM (M=43904, K=192, N=576)
  k_gemm<0><<<dim3(QKVDIM / 64, L_TOK / 128), 256, 0, stream>>>(
      Xw, WtQ, qkv_b, CDIM, QKVDIM, QKVb, nullptr, nullptr);
  // 5. windowed flash attention
  k_attn<<<NWIN * HEADS, 256, 0, stream>>>(QKVb, BiasM, AttnO);
  // 6. proj GEMM + window reverse + roll + residual
  k_gemm<1><<<dim3(CDIM / 64, L_TOK / 128), 256, 0, stream>>>(
      AttnO, WtP, proj_b, CDIM, CDIM, nullptr, xt2, xt);
  // 7. LN2
  k_ln2<<<L_TOK / 8, 256, 0, stream>>>(xt2, n2g, n2b, Xh);
  // 8. fc1 + GELU
  k_gemm<2><<<dim3(HID / 64, L_TOK / 128), 256, 0, stream>>>(
      Xh, WtF1, fc1_b, CDIM, HID, Hmid, nullptr, nullptr);
  // 9. fc2 + residual + transpose store to (C, L)
  k_gemm<3><<<dim3(CDIM / 64, L_TOK / 128), 256, 0, stream>>>(
      Hmid, WtF2, fc2_b, HID, CDIM, nullptr, out, xt2);
}
